// TriangleMultiplication_12060268167680
// MI455X (gfx1250) — matrix operand order repaired and checked by reading, not compiled, not run on hardware
//
#include <hip/hip_runtime.h>
#include <hip/hip_bf16.h>

#define NN 512
#define NP (512 * 512)
#define DP 128

typedef __attribute__((ext_vector_type(16))) __bf16 bf16x16;
typedef __attribute__((ext_vector_type(8)))  float  floatx8;

struct Q2 { uint4 a, b; };

__device__ __forceinline__ unsigned short f2bf(float f) {
    unsigned u = __float_as_uint(f);
    unsigned r = (u + 0x7FFFu + ((u >> 16) & 1u)) >> 16;  // RNE
    return (unsigned short)r;
}
__device__ __forceinline__ float bf2f(unsigned short b) {
    return __uint_as_float(((unsigned)b) << 16);
}
__device__ __forceinline__ unsigned lds_off(const void* p) {
    return (unsigned)(uintptr_t)p;   // low 32 bits of generic LDS ptr = LDS byte addr
}

// ---- CDNA5 async global<->LDS (ASYNCcnt-tracked) ----
__device__ __forceinline__ void async_load16(unsigned lds, const void* g) {
    asm volatile("global_load_async_to_lds_b128 %0, %1, off"
                 :: "v"(lds), "v"((unsigned long long)(uintptr_t)g) : "memory");
}
__device__ __forceinline__ void async_store16(const void* g, unsigned lds) {
    asm volatile("global_store_async_from_lds_b128 %0, %1, off"
                 :: "v"((unsigned long long)(uintptr_t)g), "v"(lds) : "memory");
}
#define S_WAIT_ASYNC(imm) asm volatile("s_wait_asynccnt " #imm ::: "memory")

// A fragment (16x32 bf16, MxK): lane<16 holds K {c..c+7},{16+c..}; c=8*(lane>>4)
__device__ __forceinline__ bf16x16 frag_a(const unsigned short* rowp, int hi) {
    Q2 p;
    p.a = *(const uint4*)(rowp + hi * 8);
    p.b = *(const uint4*)(rowp + 16 + hi * 8);
    return __builtin_bit_cast(bf16x16, p);
}
// B fragment (32x16 bf16, KxN), B[k][n]=W[n][k]: lane n holds K {c..c+7},{16+c..}; c=8*(lane>>4)
__device__ __forceinline__ bf16x16 frag_b(const unsigned short* colp, int hi) {
    Q2 p;
    p.a = *(const uint4*)(colp + hi * 8);
    p.b = *(const uint4*)(colp + 16 + hi * 8);
    return __builtin_bit_cast(bf16x16, p);
}
__device__ __forceinline__ floatx8 wmma_bf16(bf16x16 a, bf16x16 b, floatx8 c) {
    return __builtin_amdgcn_wmma_f32_16x16x32_bf16(false, a, false, b, (short)0, c, false, false);
}
__device__ __forceinline__ float sigmoidf(float x) {
    return __builtin_amdgcn_rcpf(1.0f + __expf(-x));   // fast v_rcp_f32, no IEEE div chain
}

// ---------------------------------------------------------------------------
// Kernel 1: LN(z)->bf16; ab=(zl@Wp^T+bp)*sig(zl@Wg^T+bg)*mask; g=sig(z@Wgg^T+bg)
// a,b written d-major bf16 via async LDS->global stores; g natural bf16.
// ---------------------------------------------------------------------------
__global__ __launch_bounds__(256) void k1_ln_proj(
    const float* __restrict__ z, const float* __restrict__ mask,
    const float* __restrict__ w_ab_p, const float* __restrict__ b_ab_p,
    const float* __restrict__ w_ab_g, const float* __restrict__ b_ab_g,
    const float* __restrict__ w_g, const float* __restrict__ b_g,
    const float* __restrict__ ln_in_g, const float* __restrict__ ln_in_b,
    unsigned short* __restrict__ aw, unsigned short* __restrict__ bw,
    unsigned short* __restrict__ gw)
{
    extern __shared__ char smem[];
    unsigned short* wp   = (unsigned short*)smem;       // [256][128]
    unsigned short* wg   = wp   + 256 * 128;            // [256][128]
    unsigned short* wgg  = wg   + 256 * 128;            // [128][128]
    unsigned short* zraw = wgg  + 128 * 128;            // [64][136] padded
    unsigned short* zln  = zraw + 64 * 136;             // [64][136] padded
    unsigned short* abst = zln  + 64 * 136;             // [256 ch][72] padded
    unsigned short* gst  = abst + 256 * 72;             // [64][128]
    float*          mlds = (float*)(gst + 64 * 128);    // [64]

    const int t = threadIdx.x;
    for (int e = t; e < 256 * 128; e += 256) wp[e]  = f2bf(w_ab_p[e]);
    for (int e = t; e < 256 * 128; e += 256) wg[e]  = f2bf(w_ab_g[e]);
    for (int e = t; e < 128 * 128; e += 256) wgg[e] = f2bf(w_g[e]);

    const int pos0 = blockIdx.x * 64;
    if (t < 64) mlds[t] = mask[pos0 + t];

    // ---- LayerNorm: 4 threads per position, quad shfl reduce ----
    {
        const int p = t >> 2, q = t & 3;
        const float* zr = z + (size_t)(pos0 + p) * DP + q * 32;
        float v[32];
        float s = 0.f;
        #pragma unroll
        for (int k = 0; k < 32; k++) { v[k] = zr[k]; s += v[k]; }
        s += __shfl_xor(s, 1); s += __shfl_xor(s, 2);
        const float mu = s * (1.f / 128.f);
        float vs = 0.f;
        #pragma unroll
        for (int k = 0; k < 32; k++) { float d = v[k] - mu; vs += d * d; }
        vs += __shfl_xor(vs, 1); vs += __shfl_xor(vs, 2);
        const float rstd = rsqrtf(vs * (1.f / 128.f) + 1e-5f);
        #pragma unroll
        for (int k = 0; k < 32; k++) {
            const int kk = q * 32 + k;
            zraw[p * 136 + kk] = f2bf(v[k]);
            zln [p * 136 + kk] = f2bf((v[k] - mu) * rstd * ln_in_g[kk] + ln_in_b[kk]);
        }
    }
    __syncthreads();

    const int wave = t >> 5, lane = t & 31;
    const int rt = wave >> 1, half = wave & 1;   // 4 row-tiles x 2 wave-halves
    const int ln15 = lane & 15, hi = lane >> 4;

    // ---- ab tiles: (zl@Wp^T+bp)*sig(zl@Wg^T+bg)*mask, masked in-epilogue ----
    const unsigned short* arow = zln + (rt * 16 + ln15) * 136;
    for (int ct = half * 8; ct < half * 8 + 8; ++ct) {
        const int ch0 = ct * 16;
        floatx8 cp = {}; floatx8 cg = {};
        const unsigned short* bpr = wp + (ch0 + ln15) * 128;
        const unsigned short* bgr = wg + (ch0 + ln15) * 128;
        #pragma unroll
        for (int kb = 0; kb < 128; kb += 32) {
            bf16x16 a = frag_a(arow + kb, hi);
            cp = wmma_bf16(a, frag_b(bpr + kb, hi), cp);
            cg = wmma_bf16(a, frag_b(bgr + kb, hi), cg);
        }
        const int ch = ch0 + ln15;
        const float bpv = b_ab_p[ch], bgv = b_ab_g[ch];
        #pragma unroll
        for (int r = 0; r < 8; ++r) {
            const int mm = r + 8 * hi;
            const float mk = mlds[rt * 16 + mm];
            const float val = (cp[r] + bpv) * sigmoidf(cg[r] + bgv) * mk;
            abst[ch * 72 + rt * 16 + mm] = f2bf(val);
        }
    }

    // ---- gate tiles: g = sigmoid(z@Wgg^T + bg) ----
    const unsigned short* araw = zraw + (rt * 16 + ln15) * 136;
    for (int gt2 = half * 4; gt2 < half * 4 + 4; ++gt2) {
        const int ch0 = gt2 * 16;
        floatx8 c = {};
        const unsigned short* brow = wgg + (ch0 + ln15) * 128;
        #pragma unroll
        for (int kb = 0; kb < 128; kb += 32)
            c = wmma_bf16(frag_a(araw + kb, hi), frag_b(brow + kb, hi), c);
        const int ch = ch0 + ln15;
        const float bb = b_g[ch];
        #pragma unroll
        for (int r = 0; r < 8; ++r) {
            const int mm = r + 8 * hi;
            gst[(rt * 16 + mm) * 128 + ch] = f2bf(sigmoidf(c[r] + bb));
        }
    }
    __syncthreads();

    // ---- drain: a/b via async LDS->global (8 x 16B per thread-channel) ----
    {
        const int ch = t;
        const unsigned short* src = abst + ch * 72;
        unsigned short* dst = (ch < 128 ? aw + (size_t)ch * NP
                                        : bw + (size_t)(ch - 128) * NP) + pos0;
        #pragma unroll
        for (int cch = 0; cch < 8; ++cch)
            async_store16(dst + cch * 8, lds_off(src + cch * 8));

        const int p = t >> 2, q = t & 3;                 // g: vectorized copy
        const uint4* gs4 = (const uint4*)(gst + p * 128 + q * 32);
        uint4* gd4 = (uint4*)(gw + (size_t)(pos0 + p) * 128 + q * 32);
        #pragma unroll
        for (int k = 0; k < 4; ++k) gd4[k] = gs4[k];
    }
    S_WAIT_ASYNC(0x0);
}

// ---------------------------------------------------------------------------
// Kernel 2: per (16i x 16j) tile: x[.,.,d]=A_d B_d^T, async-staged tiles,
//           LN over d in LDS, out = gate * (LN(x)@Wz^T + bz).
// ---------------------------------------------------------------------------
__global__ __launch_bounds__(256) void k2_einsum_out(
    const unsigned short* __restrict__ aw, const unsigned short* __restrict__ bw,
    const unsigned short* __restrict__ gw,
    const float* __restrict__ w_z, const float* __restrict__ b_z,
    const float* __restrict__ ln_out_g, const float* __restrict__ ln_out_b,
    float* __restrict__ out)
{
    extern __shared__ char smem[];
    unsigned short* wz = (unsigned short*)smem;                        // [128][128]
    float* xs          = (float*)(smem + 128 * 128 * 2);               // [256][132]
    unsigned short* xn = (unsigned short*)((char*)xs + 256 * 132 * 4); // [256][136]
    unsigned short* stg = xn + 256 * 136;   // per-wave: 2 bufs x (A 16x40 + B 16x40)

    const int t = threadIdx.x;
    for (int e = t; e < 128 * 128; e += 256) wz[e] = f2bf(w_z[e]);

    const int i0 = blockIdx.x * 16, j0 = blockIdx.y * 16;
    const int wave = t >> 5, lane = t & 31;
    const int ln15 = lane & 15, hi = lane >> 4;

    unsigned short* wstg = stg + wave * 2560;
    unsigned short* bufA[2] = { wstg, wstg + 1280 };
    unsigned short* bufB[2] = { wstg + 640, wstg + 1920 };

    // per-lane staging role: 2 lanes per row, 2 x 16B chunks each
    const int srow = lane >> 1;
    const int scol = (lane & 1) * 8;                     // ush offset of 16B chunk

    const int d0 = wave * 16;
    const size_t abase = (size_t)i0 * NN;                // + d*NP + srow*NN + kb
    const size_t bbase = (size_t)j0 * NN;

    // prologue fill: (d0, kb=0) into buf 0
    {
        const unsigned short* ga = aw + (size_t)d0 * NP + abase + srow * NN + scol;
        const unsigned short* gb = bw + (size_t)d0 * NP + bbase + srow * NN + scol;
        unsigned la = lds_off(bufA[0] + srow * 40 + scol);
        unsigned lb = lds_off(bufB[0] + srow * 40 + scol);
        async_load16(la, ga);       async_load16(la + 32, ga + 16);
        async_load16(lb, gb);       async_load16(lb + 32, gb + 16);
    }

    int cur = 0;
    floatx8 c = {};
    #pragma unroll 2
    for (int step = 0; step < 256; ++step) {
        const int nstep = step + 1;
        if (nstep < 256) {          // prefetch next (d,kstep) tile into other buf
            const int nd = d0 + (nstep >> 4);
            const int nkb = (nstep & 15) * 32;
            const unsigned short* ga = aw + (size_t)nd * NP + abase + srow * NN + nkb + scol;
            const unsigned short* gb = bw + (size_t)nd * NP + bbase + srow * NN + nkb + scol;
            unsigned la = lds_off(bufA[cur ^ 1] + srow * 40 + scol);
            unsigned lb = lds_off(bufB[cur ^ 1] + srow * 40 + scol);
            async_load16(la, ga);   async_load16(la + 32, ga + 16);
            async_load16(lb, gb);   async_load16(lb + 32, gb + 16);
            S_WAIT_ASYNC(0x4);      // in-order: current buf's 4 transfers done
        } else {
            S_WAIT_ASYNC(0x0);
        }
        c = wmma_bf16(frag_a(bufA[cur] + ln15 * 40, hi),
                      frag_b(bufB[cur] + ln15 * 40, hi), c);
        if ((step & 15) == 15) {    // finished K=512 for channel d
            const int d = d0 + (step >> 4);
            #pragma unroll
            for (int r = 0; r < 8; ++r) {
                const int p = (r + 8 * hi) * 16 + ln15;  // p = i_loc*16 + j_loc
                xs[p * 132 + d] = c[r];
            }
            c = (floatx8){};
        }
        cur ^= 1;
    }
    __syncthreads();

    // ---- output LayerNorm over d: single fused pass, float4 LDS reads ----
    {
        const int p = t;
        const float4* xr = (const float4*)(xs + p * 132);     // 32 x float4 = 128 ch
        float s = 0.f, s2 = 0.f;
        #pragma unroll 8
        for (int k4 = 0; k4 < 32; ++k4) {
            float4 v = xr[k4];
            s  += v.x + v.y + v.z + v.w;
            s2 += v.x * v.x + v.y * v.y + v.z * v.z + v.w * v.w;
        }
        const float mu = s * (1.f / 128.f);
        const float var = s2 * (1.f / 128.f) - mu * mu;
        const float rstd = rsqrtf(var + 1e-5f);
        for (int k = 0; k < 128; k += 2) {
            const float a0 = (xs[p * 132 + k]     - mu) * rstd * ln_out_g[k]     + ln_out_b[k];
            const float a1 = (xs[p * 132 + k + 1] - mu) * rstd * ln_out_g[k + 1] + ln_out_b[k + 1];
            *(unsigned*)(xn + p * 136 + k) = (unsigned)f2bf(a0) | ((unsigned)f2bf(a1) << 16);
        }
    }
    __syncthreads();

    // ---- out GEMM: [256 pos x 128] = xn @ Wz^T, gated, f32 out ----
    for (int mt = wave; mt < 16; mt += 8) {
        const unsigned short* arow = xn + (mt * 16 + ln15) * 136;
        for (int nt = 0; nt < 8; ++nt) {
            const int ch = nt * 16 + ln15;
            const unsigned short* brow = wz + ch * 128;
            floatx8 cc = {};
            #pragma unroll
            for (int kb = 0; kb < 128; kb += 32)
                cc = wmma_bf16(frag_a(arow + kb, hi), frag_b(brow + kb, hi), cc);
            const float bz = b_z[ch];
            #pragma unroll
            for (int r = 0; r < 8; ++r) {
                const int p = mt * 16 + r + 8 * hi;
                const int il = p >> 4, jl = p & 15;
                const size_t grow = (size_t)(i0 + il) * NN + (size_t)(j0 + jl);
                out[grow * 128 + ch] = (cc[r] + bz) * bf2f(gw[grow * 128 + ch]);
            }
        }
    }
}

extern "C" void kernel_launch(void* const* d_in, const int* in_sizes, int n_in,
                              void* d_out, int out_size, void* d_ws, size_t ws_size,
                              hipStream_t stream) {
    const float* z       = (const float*)d_in[0];
    const float* mask    = (const float*)d_in[1];
    const float* w_ab_p  = (const float*)d_in[2];
    const float* b_ab_p  = (const float*)d_in[3];
    const float* w_ab_g  = (const float*)d_in[4];
    const float* b_ab_g  = (const float*)d_in[5];
    const float* w_g     = (const float*)d_in[6];
    const float* b_g     = (const float*)d_in[7];
    const float* w_z     = (const float*)d_in[8];
    const float* b_z     = (const float*)d_in[9];
    const float* ln_in_g = (const float*)d_in[10];
    const float* ln_in_b = (const float*)d_in[11];
    const float* ln_out_g= (const float*)d_in[12];
    const float* ln_out_b= (const float*)d_in[13];

    unsigned short* aw = (unsigned short*)d_ws;            // [128][512][512] bf16
    unsigned short* bw = aw + (size_t)128 * NP;            // [128][512][512] bf16
    unsigned short* gw = bw + (size_t)128 * NP;            // [512*512][128]  bf16

    const size_t lds1 = (size_t)(256*128 + 256*128 + 128*128 + 64*136 + 64*136
                                 + 256*72 + 64*128) * 2 + 64 * 4;  // 252160 B
    k1_ln_proj<<<NP / 64, 256, lds1, stream>>>(
        z, mask, w_ab_p, b_ab_p, w_ab_g, b_ab_g, w_g, b_g,
        ln_in_g, ln_in_b, aw, bw, gw);

    const size_t lds2 = (size_t)128*128*2 + (size_t)256*132*4
                      + (size_t)256*136*2 + (size_t)8*2560*2;      // 278528 B
    k2_einsum_out<<<dim3(32, 32), 256, lds2, stream>>>(
        aw, bw, gw, w_z, b_z, ln_out_g, ln_out_b, (float*)d_out);
}
